// Attention_9028021256516
// MI455X (gfx1250) — matrix operand order repaired and checked by reading, compiled, not run on hardware
//
#include <hip/hip_runtime.h>
#include <hip/hip_bf16.h>

typedef __attribute__((ext_vector_type(16))) _Float16 v16h;
typedef __attribute__((ext_vector_type(8)))  float    v8f;

#define Bsz 16
#define Fsz 128
#define Ssz 256
#define Hsz 256
#define Ksz 256

__device__ __forceinline__ float fast_tanh(float x) {
#if __has_builtin(__builtin_amdgcn_tanhf)
    return __builtin_amdgcn_tanhf(x);
#else
    return tanhf(x);
#endif
}

// out = A @ W^T  where A is [M,256] row-major fp32, W is [256,256] row-major fp32.
// TRANSPOSE_OUT == false : out[m*256 + n]                     (used for q: [B*F, H])
// TRANSPOSE_OUT == true  : out[(m>>8)*65536 + n*256 + (m&255)] (used for c: [B, H, S])
// One wave computes one 16x16 tile of the output; K-loop of 8 x v_wmma_f32_16x16x32_f16.
template <bool TRANSPOSE_OUT>
__global__ __launch_bounds__(256) void proj_wmma(
    const float* __restrict__ A,
    const float* __restrict__ W,
    float* __restrict__ out)
{
    const int lane  = threadIdx.x & 31;
    const int wave  = (blockIdx.x * (blockDim.x >> 5)) + (threadIdx.x >> 5);
    const int tM    = wave >> 4;          // 16 tiles along N (=256/16)
    const int tN    = wave & 15;

    const int arow_i = tM * 16 + (lane & 15);   // A row held by this lane
    const int wrow_i = tN * 16 + (lane & 15);   // W row (output column) held by this lane
    const int aKoff  = (lane < 16) ? 0 : 8;     // A frag: elems 0..7 -> K+aKoff+0..7, elems 8..15 -> K+16+aKoff+0..7
    const int bKoff  = (lane < 16) ? 0 : 8;     // B frag: elems 0..7 -> K+bKoff+0..7, elems 8..15 -> K+16+bKoff+0..7

    const float* __restrict__ arow = A + arow_i * Ksz;
    const float* __restrict__ wrow = W + wrow_i * Ksz;

    v8f acc = {};
    #pragma unroll
    for (int k0 = 0; k0 < Ksz; k0 += 32) {
        v16h af, bf;
        // ---- A fragment (16x32 f16, ISA layout) ----
        float4 x0 = *(const float4*)(arow + k0 + aKoff);
        float4 x1 = *(const float4*)(arow + k0 + aKoff + 4);
        float4 x2 = *(const float4*)(arow + k0 + 16 + aKoff);
        float4 x3 = *(const float4*)(arow + k0 + 16 + aKoff + 4);
        af[0]=(_Float16)x0.x;  af[1]=(_Float16)x0.y;  af[2]=(_Float16)x0.z;  af[3]=(_Float16)x0.w;
        af[4]=(_Float16)x1.x;  af[5]=(_Float16)x1.y;  af[6]=(_Float16)x1.z;  af[7]=(_Float16)x1.w;
        af[8]=(_Float16)x2.x;  af[9]=(_Float16)x2.y;  af[10]=(_Float16)x2.z; af[11]=(_Float16)x2.w;
        af[12]=(_Float16)x3.x; af[13]=(_Float16)x3.y; af[14]=(_Float16)x3.z; af[15]=(_Float16)x3.w;
        // ---- B fragment (32x16 f16): B[k][n] = W[n][k] ----
        float4 y0 = *(const float4*)(wrow + k0 + bKoff);
        float4 y1 = *(const float4*)(wrow + k0 + bKoff + 4);
        float4 y2 = *(const float4*)(wrow + k0 + 16 + bKoff);
        float4 y3 = *(const float4*)(wrow + k0 + 16 + bKoff + 4);
        bf[0]=(_Float16)y0.x;  bf[1]=(_Float16)y0.y;  bf[2]=(_Float16)y0.z;  bf[3]=(_Float16)y0.w;
        bf[4]=(_Float16)y1.x;  bf[5]=(_Float16)y1.y;  bf[6]=(_Float16)y1.z;  bf[7]=(_Float16)y1.w;
        bf[8]=(_Float16)y2.x;  bf[9]=(_Float16)y2.y;  bf[10]=(_Float16)y2.z; bf[11]=(_Float16)y2.w;
        bf[12]=(_Float16)y3.x; bf[13]=(_Float16)y3.y; bf[14]=(_Float16)y3.z; bf[15]=(_Float16)y3.w;

        acc = __builtin_amdgcn_wmma_f32_16x16x32_f16(
                  /*neg_a=*/false, af, /*neg_b=*/false, bf,
                  /*c_mod=*/(short)0, acc, /*reuse_a=*/false, /*reuse_b=*/false);
    }

    // C/D layout: VGPR r -> M = r (lanes 0-15) or r+8 (lanes 16-31), N = lane&15
    const int rbase = tM * 16 + ((lane < 16) ? 0 : 8);
    const int col   = tN * 16 + (lane & 15);
    #pragma unroll
    for (int r = 0; r < 8; ++r) {
        const int row = rbase + r;
        if (TRANSPOSE_OUT) {
            out[(row >> 8) * (Hsz * Ssz) + col * Ssz + (row & 255)] = acc[r];
        } else {
            out[row * Hsz + col] = acc[r];
        }
    }
}

// One block per (b,f). energy[s] = sum_h tanh(q[bf,h] + cT[b,h,s]) * v[h], then softmax over s.
__global__ __launch_bounds__(256) void energy_softmax(
    const float* __restrict__ q,    // [B*F, H]
    const float* __restrict__ cT,   // [B, H, S]
    const float* __restrict__ v,    // [H]
    float* __restrict__ out)        // [B, F, S]
{
    __shared__ float qrow[Hsz];
    __shared__ float vv[Hsz];
    __shared__ float red[Ssz];

    const int bf = blockIdx.x;
    const int b  = bf >> 7;         // F = 128
    const int t  = threadIdx.x;     // s index

    qrow[t] = q[bf * Hsz + t];
    vv[t]   = v[t];
    __syncthreads();

    const float* __restrict__ cb = cT + b * (Hsz * Ssz);
    float e = 0.0f;
    #pragma unroll 8
    for (int h = 0; h < Hsz; ++h) {
        e += fast_tanh(qrow[h] + cb[h * Ssz + t]) * vv[h];   // lane-contiguous global read
    }

    // softmax over the 256 threads (s axis)
    red[t] = e;
    __syncthreads();
    #pragma unroll
    for (int off = 128; off > 0; off >>= 1) {
        if (t < off) red[t] = fmaxf(red[t], red[t + off]);
        __syncthreads();
    }
    const float m = red[0];
    __syncthreads();
    const float ex = __expf(e - m);
    red[t] = ex;
    __syncthreads();
    #pragma unroll
    for (int off = 128; off > 0; off >>= 1) {
        if (t < off) red[t] += red[t + off];
        __syncthreads();
    }
    out[bf * Ssz + t] = ex / red[0];
}

extern "C" void kernel_launch(void* const* d_in, const int* in_sizes, int n_in,
                              void* d_out, int out_size, void* d_ws, size_t ws_size,
                              hipStream_t stream) {
    const float* query   = (const float*)d_in[0];  // [16,128,256]
    const float* context = (const float*)d_in[1];  // [16,256,256]
    const float* W_q     = (const float*)d_in[2];  // [256,256]
    const float* W_c     = (const float*)d_in[3];  // [256,256]
    const float* v       = (const float*)d_in[4];  // [256]

    float* q_ws  = (float*)d_ws;                   // 2048*256 f32 = 2 MB
    float* cT_ws = q_ws + (Bsz * Fsz * Hsz);       // 4096*256 f32 = 4 MB

    // q projection: M = B*F = 2048 rows -> 2048 waves -> 256 blocks of 8 waves
    proj_wmma<false><<<(Bsz * Fsz) / 8, 256, 0, stream>>>(query, W_q, q_ws);
    // c projection (stored transposed [B,H,S]): M = B*S = 4096 -> 512 blocks
    proj_wmma<true><<<(Bsz * Ssz) / 8, 256, 0, stream>>>(context, W_c, cT_ws);
    // fused energy + softmax: one block per (b,f)
    energy_softmax<<<Bsz * Fsz, 256, 0, stream>>>(q_ws, cT_ws, v, (float*)d_out);
}
